// CrossEncoderReranker_89232240542409
// MI455X (gfx1250) — matrix operand order repaired and checked by reading, not compiled, not run on hardware
//
#include <hip/hip_runtime.h>

typedef _Float16 v16h __attribute__((ext_vector_type(16)));
typedef _Float16 v8h  __attribute__((ext_vector_type(8)));
typedef _Float16 v4h  __attribute__((ext_vector_type(4)));
typedef float    v8f  __attribute__((ext_vector_type(8)));

namespace {
constexpr int NM      = 4096;        // mentions
constexpr int Hdim    = 768;         // hidden
constexpr int TWO_H   = 1536;        // 2H (GEMM K)
constexpr int TOTALP  = 131072;      // ragged pairs (GEMM M)
constexpr int NOUTC   = 65;          // MAXK + 1
constexpr int KT      = 48;          // K tiles of 32 over 1536
constexpr int CTs     = 48;          // col tiles of 16 over 768
constexpr int MT      = 4;           // M tiles of 16 per workgroup
constexpr int ROWT    = 16 * MT;     // 64 rows per workgroup
constexpr int BLOCK   = 512;         // 16 wave32s (4 per SIMD)
constexpr int CT_PER_WAVE = CTs / (BLOCK / 32);   // 3
// LDS A-tile: kt-major, fragment-permuted. Row stride 40 halfs (80 B) ->
// per-pass DS bank starts are 20*m mod 64 = perfect 4-dword tiling.
constexpr int ARST    = 40;                        // halfs per row (32 + 8 pad)
constexpr int KTBLK   = 64 * ARST;                 // halfs per kt block (5120 B)
constexpr float FAISS_WEIGHT = 0.5f;
}

// ---------------------------------------------------------------------------
// Pre-pass 1: W1 (fp32, [1536 x 768] row-major) -> f16 in WMMA B-fragment
// order: W1p[((ct*KT + kt)*32 + lane)*16 + j] = W1[kt*32 + (lane>>4)*8 + (j&7) + (j>>3)*16][ct*16 + (lane&15)]
// Matches ISA 7.12.2 f16 B 32x16 layout: lanes 0-15 hold K=0-7,16-23, lanes 16-31 K=8-15,24-31.
// ---------------------------------------------------------------------------
__global__ void ce_pack_w1(const float* __restrict__ W1, _Float16* __restrict__ W1p) {
    int idx = blockIdx.x * 256 + threadIdx.x;          // half-element index
    if (idx >= CTs * KT * 32 * 16) return;
    int j    = idx & 15;
    int lane = (idx >> 4) & 31;
    int rem  = idx >> 9;
    int kt   = rem % KT;
    int ct   = rem / KT;
    int k    = kt * 32 + (lane >> 4) * 8 + (j & 7) + (j >> 3) * 16;
    int col  = ct * 16 + (lane & 15);
    W1p[idx] = (_Float16)W1[(size_t)k * Hdim + col];
}

// ---------------------------------------------------------------------------
// Pre-pass 2: out[n][k] = 0.5*prior[n][k] for k<64 ; out[n][64] = nota_bias.
// ---------------------------------------------------------------------------
__global__ void ce_init_out(const float* __restrict__ prior,
                            const float* __restrict__ nota,
                            float* __restrict__ out) {
    int idx = blockIdx.x * 256 + threadIdx.x;
    if (idx >= NM * NOUTC) return;
    int n = idx / NOUTC;
    int k = idx - n * NOUTC;
    out[idx] = (k < 64) ? FAISS_WEIGHT * prior[n * 64 + k] : nota[0];
}

// ---------------------------------------------------------------------------
// Main fused kernel: 64 ragged rows per workgroup (4 M-tiles of 16).
// LDS A-tile is stored kt-major with each row's 32 K-values permuted into
// WMMA A-fragment order [K0-7 | K16-23 | K8-15 | K24-31], so a lane's whole
// fragment is one contiguous 32-B read at (row*80 + hi*32) — no shuffles,
// all M-tile/kt offsets fit the 16-bit DS immediate field.
// ---------------------------------------------------------------------------
__global__ __launch_bounds__(BLOCK)
void ce_main(const float* __restrict__ me, const float* __restrict__ ce,
             const _Float16* __restrict__ W1p, const float* __restrict__ b1,
             const float* __restrict__ W2, const float* __restrict__ b2,
             const int* __restrict__ midx, const int* __restrict__ cidx,
             float* __restrict__ out)
{
    extern __shared__ char smem[];
    _Float16* Atile    = (_Float16*)smem;                               // [KT][64][ARST]
    float*    scoreBuf = (float*)(smem + (size_t)KT * KTBLK * sizeof(_Float16));

    const int tid = threadIdx.x;
    const int t0  = blockIdx.x * ROWT;

    if (tid < ROWT) scoreBuf[tid] = 0.0f;

    // Gather + fp32->f16 convert + fragment-permute 64 rows of x into LDS.
    constexpr int CH = TWO_H / 4;                       // 384 float4 chunks per row
    for (int i = tid; i < ROWT * CH; i += BLOCK) {
        int row = i / CH;
        int c4  = (i - row * CH) * 4;                   // K index, multiple of 4
        int t   = t0 + row;
        const float* src = (c4 < Hdim)
            ? me + (size_t)midx[t] * Hdim + c4
            : ce + (size_t)t       * Hdim + (c4 - Hdim);
        float4 v = *(const float4*)src;
        v4h hh;
        hh.x = (_Float16)v.x; hh.y = (_Float16)v.y;
        hh.z = (_Float16)v.z; hh.w = (_Float16)v.w;
        int kt  = c4 >> 5;
        int Kin = c4 & 31;
        // fragment order within 32-K block: [0-7 | 16-23 | 8-15 | 24-31]
        int pos = (Kin & 7) | (((Kin >> 3) & 1) << 4) | (((Kin >> 4) & 1) << 3);
        *(v4h*)(Atile + (size_t)kt * KTBLK + (size_t)row * ARST + pos) = hh;
    }
    __syncthreads();

    const int lane = tid & 31;
    const int wave = __builtin_amdgcn_readfirstlane(tid >> 5);  // wave-uniform SGPR
    const int m    = lane & 15;
    const int hi   = lane >> 4;

    const v16h* __restrict__ Bfrag = (const v16h*)W1p;

    float sacc[MT][8];
    #pragma unroll
    for (int mt = 0; mt < MT; ++mt)
        #pragma unroll
        for (int r = 0; r < 8; ++r) sacc[mt][r] = 0.0f;

    // Per-lane A-fragment base: row m of M-tile 0; hi selects 32-B half.
    const _Float16* abase = Atile + (size_t)m * ARST + hi * 16;

    const int ct0 = wave * CT_PER_WAVE;
    for (int ct = ct0; ct < ct0 + CT_PER_WAVE; ++ct) {
        v8f c[MT];
        #pragma unroll
        for (int mt = 0; mt < MT; ++mt) c[mt] = (v8f){};

        const v16h* bp = Bfrag + (size_t)ct * KT * 32 + lane;
        #pragma unroll 2
        for (int kt = 0; kt < KT; ++kt) {
            v16h b = bp[(size_t)kt * 32];               // 2x global_load_b128 (L2)
            const _Float16* ak = abase + (size_t)kt * KTBLK;
            #pragma unroll
            for (int mt = 0; mt < MT; ++mt) {
                v16h a = *(const v16h*)(ak + mt * (16 * ARST));  // 2x ds_load_b128, imm offs
                c[mt] = __builtin_amdgcn_wmma_f32_16x16x32_f16(false, a, false, b,
                                                               (short)0, c[mt], false, false);
            }
        }
        // Fused epilogue: +b1, relu, dot with W2 column chunk.
        const int col  = ct * 16 + m;                   // C/D layout: N = lane&15
        const float w2 = W2[col];
        const float bb = b1[col];
        #pragma unroll
        for (int mt = 0; mt < MT; ++mt) {
            #pragma unroll
            for (int r = 0; r < 8; ++r) {
                float hv = c[mt][r] + bb;
                hv = hv > 0.0f ? hv : 0.0f;
                sacc[mt][r] += hv * w2;
            }
        }
    }

    // C/D row mapping: VGPR r holds M = mt*16 + r + hi*8. Reduce via LDS atomics.
    #pragma unroll
    for (int mt = 0; mt < MT; ++mt)
        #pragma unroll
        for (int r = 0; r < 8; ++r)
            atomicAdd(&scoreBuf[mt * 16 + hi * 8 + r], sacc[mt][r]);   // ds_add_f32

    __syncthreads();

    if (tid < ROWT) {
        int t = t0 + tid;
        float s = scoreBuf[tid] + b2[0];
        // unique (mention, col) per pair -> race-free plain RMW on prior base
        out[(size_t)midx[t] * NOUTC + cidx[t]] += s;
    }
}

// ---------------------------------------------------------------------------
extern "C" void kernel_launch(void* const* d_in, const int* in_sizes, int n_in,
                              void* d_out, int out_size, void* d_ws, size_t ws_size,
                              hipStream_t stream) {
    (void)in_sizes; (void)n_in; (void)out_size;
    const float* me   = (const float*)d_in[0];   // mention_embs   [4096, 768]
    const float* ce   = (const float*)d_in[1];   // candidate_embs [131072, 768]
    const float* W1   = (const float*)d_in[2];   // [1536, 768]
    const float* b1   = (const float*)d_in[3];   // [768]
    const float* W2   = (const float*)d_in[4];   // [768, 1]
    const float* b2   = (const float*)d_in[5];   // [1]
    const float* pr   = (const float*)d_in[6];   // faiss_prior [4096, 64]
    const float* nb   = (const float*)d_in[7];   // nota_bias scalar
    const int*   midx = (const int*)d_in[8];     // [131072]
    const int*   cidx = (const int*)d_in[9];     // [131072]
    float* out = (float*)d_out;                  // [4096, 65]

    _Float16* W1p = (_Float16*)d_ws;             // needs 2,359,296 B of scratch
    if (ws_size < (size_t)CTs * KT * 32 * 16 * sizeof(_Float16)) return;

    const int packN = CTs * KT * 32 * 16;
    ce_pack_w1<<<(packN + 255) / 256, 256, 0, stream>>>(W1, W1p);
    ce_init_out<<<(NM * NOUTC + 255) / 256, 256, 0, stream>>>(pr, nb, out);

    const size_t smem = (size_t)KT * KTBLK * sizeof(_Float16)
                      + (size_t)ROWT * sizeof(float);   // ~240.3 KB of 320 KB WGP LDS
    ce_main<<<TOTALP / ROWT, BLOCK, smem, stream>>>(me, ce, W1p, b1, W2, b2,
                                                    midx, cidx, out);
}
